// SEMAttn_86973087744457
// MI455X (gfx1250) — matrix operand order repaired and checked by reading, not compiled, not run on hardware
//
#include <hip/hip_runtime.h>
#include <hip/hip_bf16.h>
#include <cmath>

// ---------------------------------------------------------------------------
// Problem constants (from the reference): 64 spectral modes x 1024 mesh points
// = 65536 "tokens"; IDIM=256, H=8, DH=64, NDIM=2, E=H*DH*NDIM reshaped so each
// token has q,k,v of shape (H=8, 128); attention is over heads (8x8 scores);
// output GEMM is attn(*,1024) @ Wout(1024,256).
// ---------------------------------------------------------------------------

typedef __attribute__((ext_vector_type(16))) __bf16 bf16x16;
typedef __attribute__((ext_vector_type(8)))  float  f32x8;

#define NTOK      65536
#define TOK_TILE  16
#define IDIM_     256
#define PROJ_N    512      // cols of each per-dim projection (H*DH)
#define ODIM_     256
#define EPS_      1e-5f

// ws layout (bf16 elements), all pre-swizzled into WMMA B-fragment order:
//  WQ: [n(2)][ct(32)][kc(8)][lane(32)][e(16)]  = 262144
//  WK: +262144, WV: +262144
//  WO: [n(2)][ct(16)][kc(16)][lane(32)][e(16)] = 262144
#define WS_WQ 0
#define WS_WK 262144
#define WS_WV 524288
#define WS_WO 786432
#define WS_TOTAL 1048576   // bf16 elements -> 2 MB

// ---------------------------------------------------------------------------
// Prep: convert f32 weights -> bf16 in B-fragment order.
// 16-bit B 32x16 layout (ISA 7.12.2): lane L -> N = L%16,
//   K = (e/8)*16 + ((L>=16)?8:0) + (e&7)   for fragment element e in [0,16).
// ---------------------------------------------------------------------------
__global__ __launch_bounds__(256)
void semattn_prep_weights(const float* __restrict__ WQ,
                          const float* __restrict__ WK,
                          const float* __restrict__ WV,
                          const float* __restrict__ Wout,
                          __bf16* __restrict__ ws)
{
    int idx = blockIdx.x * 256 + threadIdx.x;
    if (idx >= WS_TOTAL) return;

    if (idx < WS_WO) {
        // ---- projection weights (2,256,512) row-major per matrix ----
        int mat = idx / 262144;          // 0=Q 1=K 2=V
        int rem = idx % 262144;
        int n   = rem / 131072;  rem %= 131072;
        int ct  = rem / 4096;    rem %= 4096;     // 32 col-tiles of 16
        int kc  = rem / 512;     rem %= 512;      // 8 K-chunks of 32
        int L   = rem / 16;
        int e   = rem % 16;
        int kk  = (e >> 3) * 16 + ((L >= 16) ? 8 : 0) + (e & 7);
        int kg  = kc * 32 + kk;                   // 0..255
        int col = ct * 16 + (L & 15);             // 0..511
        const float* W = (mat == 0) ? WQ : ((mat == 1) ? WK : WV);
        ws[idx] = (__bf16)W[((size_t)n * IDIM_ + kg) * PROJ_N + col];
    } else {
        // ---- Wout (1024,256): row = h*128 + n*64 + d ----
        int rem = idx - WS_WO;
        int n   = rem / 131072;  rem %= 131072;
        int ct  = rem / 8192;    rem %= 8192;     // 16 col-tiles of 16
        int kc  = rem / 512;     rem %= 512;      // 16 K-chunks of 32
        int L   = rem / 16;
        int e   = rem % 16;
        int kk  = (e >> 3) * 16 + ((L >= 16) ? 8 : 0) + (e & 7);
        int c   = kc * 32 + kk;                   // chunk col 0..511 = h*64+d
        int h   = c >> 6, d = c & 63;
        int col = ct * 16 + (L & 15);
        ws[idx] = (__bf16)Wout[((size_t)(h * 128 + n * 64 + d)) * ODIM_ + col];
    }
}

// ---------------------------------------------------------------------------
// Fused kernel: 128 threads (4 waves), 16 tokens per workgroup.
//   phase 0: stage X tile (16x256) in LDS in WMMA A-fragment order (bf16)
//   phase 1 (n=0,1): WMMA q/k raw -> LDS; VALU LN+rotate in place;
//                    VALU accumulate 8x8 scores into sP
//   phase 2: softmax over 8 (per token,head)
//   phase 3 (n=0,1): WMMA v raw -> LDS; VALU attn = P@v (fragment-order out);
//                    WMMA out-accumulate vs pre-swizzled Wout
//   phase 4: +bout, store f32 tile
// 16-bit A 16x32 layout (ISA 7.12.2): lane L -> M = L%16,
//   K = (e/2/4)*16 + ((L>=16)?8:0) + ((e/2)%4)*2 + (e&1)
// ---------------------------------------------------------------------------
__global__ __launch_bounds__(128)
void semattn_fused(const float* __restrict__ nodal,
                   const float* __restrict__ coords,
                   const float* __restrict__ biasQ,
                   const float* __restrict__ biasK,
                   const float* __restrict__ lnQg, const float* __restrict__ lnQb,
                   const float* __restrict__ lnKg, const float* __restrict__ lnKb,
                   const float* __restrict__ bout,
                   const __bf16* __restrict__ ws,
                   float* __restrict__ out)
{
    __shared__ __bf16 sX[8 * 512];        // X tile, A-frag order      (8 KB)
    __shared__ __bf16 sQ[16 * 512];       // q_n rot / v_n raw         (16 KB)
    __shared__ __bf16 sK[16 * 512];       // k_n rot                   (16 KB)
    __shared__ __bf16 sAtt[16 * 512];     // attn chunk, A-frag order  (16 KB)
    __shared__ float  sP[16 * 64];        // scores / probs 16x(8x8)   (4 KB)
    __shared__ float  sC[16 * 2];         // coords per token

    const int tid  = threadIdx.x;
    const int lane = tid & 31;
    const int wave = tid >> 5;
    const int tok0 = blockIdx.x * TOK_TILE;

    // ---------------- phase 0: stage inputs ----------------
    for (int i = 0; i < 32; ++i) {
        int idx = tid + i * 128;                  // 0..4095 (16 x 256)
        int t = idx >> 8, c = idx & 255;          // coalesced over c
        float v = nodal[((size_t)(tok0 + t)) * IDIM_ + c];
        int kc = c >> 5, kk = c & 31;
        int grp = kk >> 4, rem = kk & 15;
        int hi  = (rem >= 8) ? 1 : 0;
        int rr  = (rem & 7) >> 1, p = rem & 1;
        sX[kc * 512 + (t + hi * 16) * 16 + (grp * 4 + rr) * 2 + p] = (__bf16)v;
    }
    if (tid < 32) sC[tid] = coords[(size_t)tok0 * 2 + tid];
    for (int i = tid; i < 16 * 64; i += 128) sP[i] = 0.f;
    __syncthreads();

    // ---------------- phase 1: Q/K projections + LN + rotation + scores ----
    for (int n = 0; n < 2; ++n) {
        // 64 GEMM col-tile jobs: j<32 -> Q tile j ; j>=32 -> K tile j-32
        for (int j = wave; j < 64; j += 4) {
            int  isK = j >> 5;
            int  ct  = j & 31;
            const __bf16* wb = ws + (isK ? WS_WK : WS_WQ) + n * 131072 + ct * 4096;
            __builtin_prefetch(wb + 4096, 0, 1);   // warm next weight tile in L2
            f32x8 acc = {};
            for (int kc = 0; kc < 8; ++kc) {
                bf16x16 a = *(const bf16x16*)&sX[kc * 512 + lane * 16];
                bf16x16 b = *(const bf16x16*)&wb[kc * 512 + lane * 16];
                acc = __builtin_amdgcn_wmma_f32_16x16x32_bf16(
                          false, a, false, b, (short)0, acc, false, false);
            }
            __bf16* dst = isK ? sK : sQ;
            int col = ct * 16 + (lane & 15);
            int mb  = (lane >= 16) ? 8 : 0;
            for (int r = 0; r < 8; ++r)
                dst[(mb + r) * 512 + col] = (__bf16)acc[r];
        }
        __syncthreads();

        // LN + rotation, one (token,head) per thread, for both Q and K
        {
            int t = tid >> 3, h = tid & 7;
            float cn = sC[t * 2 + n];
            for (int pk = 0; pk < 2; ++pk) {
                __bf16* seg = (pk ? sK : sQ) + t * 512 + h * 64;
                const float* bias = (pk ? biasK : biasQ) + (n * 8 + h) * 64;
                const float* gam  = (pk ? lnKg : lnQg) + n * 64;
                const float* bet  = (pk ? lnKb : lnQb) + n * 64;
                float mu = 0.f, s2 = 0.f;
                for (int d = 0; d < 64; ++d) {
                    float x = (float)seg[d] + bias[d];
                    mu += x; s2 += x * x;
                }
                mu *= (1.f / 64.f);
                float inv = rsqrtf(s2 * (1.f / 64.f) - mu * mu + EPS_);
                for (int d = 0; d < 32; ++d) {
                    float x0 = (float)seg[d]      + bias[d];
                    float x1 = (float)seg[d + 32] + bias[d + 32];
                    float f = (x0 - mu) * inv * gam[d]      + bet[d];
                    float g = (x1 - mu) * inv * gam[d + 32] + bet[d + 32];
                    float ang = cn * (float)d;
                    float ck = cosf(ang), sk = sinf(ang);
                    float fr = ck * f - sk * g;
                    float gr = sk * f + ck * g;
                    seg[d]      = (__bf16)(ck + fr);
                    seg[d + 32] = (__bf16)(sk + gr);
                }
            }
        }
        __syncthreads();

        // scores partial: P(t,h1,h2) += sum_d q(t,h1,d) * k(t,h2,d)
        for (int i = 0; i < 8; ++i) {
            int idx = tid * 8 + i;                 // 0..1023 = t*64 + h1*8 + h2
            int t = idx >> 6, h1 = (idx >> 3) & 7, h2 = idx & 7;
            const __bf16* qr = sQ + t * 512 + h1 * 64;
            const __bf16* kr = sK + t * 512 + h2 * 64;
            float s = 0.f;
            for (int d = 0; d < 64; ++d) s += (float)qr[d] * (float)kr[d];
            sP[idx] += s;
        }
        __syncthreads();
    }

    // ---------------- phase 2: softmax over 8 keys (per token,head) --------
    {
        int t = tid >> 3, h = tid & 7;
        float* row = sP + t * 64 + h * 8;
        const float scale = 0.08838834764831845f;   // 1/sqrt(128)
        float m = -INFINITY;
        for (int j = 0; j < 8; ++j) { row[j] *= scale; m = fmaxf(m, row[j]); }
        float sum = 0.f;
        for (int j = 0; j < 8; ++j) { float e = expf(row[j] - m); row[j] = e; sum += e; }
        float r = 1.f / sum;
        for (int j = 0; j < 8; ++j) row[j] *= r;
    }
    __syncthreads();

    // ---------------- phase 3: V projection, attn = P@V, output GEMM ------
    f32x8 oacc[4] = {{}, {}, {}, {}};              // wave owns out col-tiles 4w..4w+3
    for (int n = 0; n < 2; ++n) {
        __syncthreads();
        // v raw -> sQ (reuse)
        for (int ct = wave; ct < 32; ct += 4) {
            const __bf16* wb = ws + WS_WV + n * 131072 + ct * 4096;
            __builtin_prefetch(wb + 4096, 0, 1);
            f32x8 acc = {};
            for (int kc = 0; kc < 8; ++kc) {
                bf16x16 a = *(const bf16x16*)&sX[kc * 512 + lane * 16];
                bf16x16 b = *(const bf16x16*)&wb[kc * 512 + lane * 16];
                acc = __builtin_amdgcn_wmma_f32_16x16x32_bf16(
                          false, a, false, b, (short)0, acc, false, false);
            }
            int col = ct * 16 + (lane & 15);
            int mb  = (lane >= 16) ? 8 : 0;
            for (int r = 0; r < 8; ++r)
                sQ[(mb + r) * 512 + col] = (__bf16)acc[r];
        }
        __syncthreads();

        // attn_n(t, h*64+d) = sum_j P(t,h,j) * v_n(t, j*64+d); write A-frag order
        for (int i = 0; i < 64; ++i) {
            int idx = tid + i * 128;               // 0..8191
            int t = idx >> 9, c = idx & 511;
            int h = c >> 6, d = c & 63;
            const float*  Pr = sP + t * 64 + h * 8;
            const __bf16* vt = sQ + t * 512 + d;
            float s = 0.f;
            for (int j = 0; j < 8; ++j) s += Pr[j] * (float)vt[j * 64];
            int kc = c >> 5, kk = c & 31;
            int grp = kk >> 4, rem = kk & 15;
            int hi  = (rem >= 8) ? 1 : 0;
            int rr  = (rem & 7) >> 1, p = rem & 1;
            sAtt[kc * 512 + (t + hi * 16) * 16 + (grp * 4 + rr) * 2 + p] = (__bf16)s;
        }
        __syncthreads();

        // out tile accumulate: attn_n(16x512) @ Wout_n(512x256), K-chunks of 32
        for (int kc = 0; kc < 16; ++kc) {
            bf16x16 a = *(const bf16x16*)&sAtt[kc * 512 + lane * 16];
            for (int i = 0; i < 4; ++i) {
                int ct = wave * 4 + i;
                const __bf16* wb = ws + WS_WO + n * 131072 + ct * 8192 + kc * 512;
                bf16x16 b = *(const bf16x16*)&wb[lane * 16];
                oacc[i] = __builtin_amdgcn_wmma_f32_16x16x32_bf16(
                              false, a, false, b, (short)0, oacc[i], false, false);
            }
        }
    }

    // ---------------- phase 4: bias + store ----------------
    {
        int mb = (lane >= 16) ? 8 : 0;
        int cl = lane & 15;
        for (int i = 0; i < 4; ++i) {
            int col = (wave * 4 + i) * 16 + cl;
            float bo = bout[col];
            for (int r = 0; r < 8; ++r)
                out[((size_t)(tok0 + mb + r)) * ODIM_ + col] = oacc[i][r] + bo;
        }
    }
}

// ---------------------------------------------------------------------------
extern "C" void kernel_launch(void* const* d_in, const int* in_sizes, int n_in,
                              void* d_out, int out_size, void* d_ws, size_t ws_size,
                              hipStream_t stream)
{
    const float* nodal  = (const float*)d_in[0];
    const float* coords = (const float*)d_in[1];
    const float* WQ     = (const float*)d_in[2];
    const float* WK     = (const float*)d_in[3];
    const float* WV     = (const float*)d_in[4];
    const float* biasQ  = (const float*)d_in[5];
    const float* biasK  = (const float*)d_in[6];
    const float* lnQg   = (const float*)d_in[7];
    const float* lnQb   = (const float*)d_in[8];
    const float* lnKg   = (const float*)d_in[9];
    const float* lnKb   = (const float*)d_in[10];
    const float* Wout   = (const float*)d_in[11];
    const float* bout   = (const float*)d_in[12];
    float*  out = (float*)d_out;
    __bf16* ws  = (__bf16*)d_ws;      // needs 2 MB

    semattn_prep_weights<<<WS_TOTAL / 256, 256, 0, stream>>>(WQ, WK, WV, Wout, ws);
    semattn_fused<<<NTOK / TOK_TILE, 128, 0, stream>>>(
        nodal, coords, biasQ, biasK, lnQg, lnQb, lnKg, lnKb, bout, ws, out);
}
